// QuantLinearW8A8_20675972563263
// MI455X (gfx1250) — 8-bit matrix operand order repaired and checked by reading, compiled, not run on hardware
//
#include <hip/hip_runtime.h>
#include <stdint.h>

typedef int v8i __attribute__((ext_vector_type(8)));
typedef int v4i __attribute__((ext_vector_type(4)));
typedef int v2i __attribute__((ext_vector_type(2)));

union U8 {
    v8i v;
    v4i q[2];
    v2i d[4];
};

__device__ __forceinline__ unsigned lds_addr_of(const void* p) {
    return (unsigned)(size_t)p;
}

__device__ __forceinline__ void async_ld_b128(unsigned lds_off, const void* gsrc) {
    asm volatile("global_load_async_to_lds_b128 %0, %1, off"
                 :: "v"(lds_off), "v"((unsigned long long)(size_t)gsrc)
                 : "memory");
}

__device__ __forceinline__ void wait_async_le2() {
#if __has_builtin(__builtin_amdgcn_s_wait_asynccnt)
    __builtin_amdgcn_s_wait_asynccnt(2);
#else
    asm volatile("s_wait_asynccnt 0x2" ::: "memory");
#endif
}

__device__ __forceinline__ void wait_async_le0() {
#if __has_builtin(__builtin_amdgcn_s_wait_asynccnt)
    __builtin_amdgcn_s_wait_asynccnt(0);
#else
    asm volatile("s_wait_asynccnt 0x0" ::: "memory");
#endif
}

__global__ void ws_init_kernel(unsigned* w) {
    if (threadIdx.x == 0 && blockIdx.x == 0) w[0] = 0u;
}

__global__ __launch_bounds__(256)
void absmax_kernel(const float4* __restrict__ x, unsigned* __restrict__ out, long long n4) {
    __shared__ float red[256];
    float m = 0.0f;
    long long i      = (long long)blockIdx.x * blockDim.x + threadIdx.x;
    long long stride = (long long)gridDim.x * blockDim.x;
    for (; i < n4; i += stride) {
        float4 v = x[i];
        m = fmaxf(m, fmaxf(fmaxf(fabsf(v.x), fabsf(v.y)),
                           fmaxf(fabsf(v.z), fabsf(v.w))));
    }
    red[threadIdx.x] = m;
    __syncthreads();
    for (int s = 128; s > 0; s >>= 1) {
        if ((int)threadIdx.x < s)
            red[threadIdx.x] = fmaxf(red[threadIdx.x], red[threadIdx.x + s]);
        __syncthreads();
    }
    if (threadIdx.x == 0)
        atomicMax(out, __float_as_uint(red[0]));
}

__global__ __launch_bounds__(256)
void quant_kernel(const float4* __restrict__ x, int* __restrict__ xq_packed,
                  const unsigned* __restrict__ absmax_bits, long long n4) {
    const float amax = __uint_as_float(*absmax_bits);
    const float s_x  = amax * (1.0f / 127.0f);
    const float inv  = (s_x > 0.0f) ? (1.0f / s_x) : 0.0f;

    long long i      = (long long)blockIdx.x * blockDim.x + threadIdx.x;
    long long stride = (long long)gridDim.x * blockDim.x;
    for (; i < n4; i += stride) {
        float4 v = x[i];
        int q0 = min(127, max(-127, __float2int_rn(v.x * inv)));
        int q1 = min(127, max(-127, __float2int_rn(v.y * inv)));
        int q2 = min(127, max(-127, __float2int_rn(v.z * inv)));
        int q3 = min(127, max(-127, __float2int_rn(v.w * inv)));
        xq_packed[i] = (q0 & 0xff) | ((q1 & 0xff) << 8) |
                       ((q2 & 0xff) << 16) | ((q3 & 0xff) << 24);
    }
}

#define LDS_STRIDE 80

__global__ __launch_bounds__(256)
void gemm_w8a8_kernel(const int8_t* __restrict__ xq,
                      const int8_t* __restrict__ wq,
                      const float* __restrict__ sw,
                      const float* __restrict__ bias,
                      const unsigned* __restrict__ absmax_bits,
                      float* __restrict__ out,
                      int Mdim, int Ndim, int Kdim) {
    __shared__ __align__(16) int8_t bsm[2][128 * LDS_STRIDE];

    const int tid  = threadIdx.x;
    const int lane = tid & 31;
    const int wave = tid >> 5;
    const int n0 = blockIdx.x * 128;
    const int m0 = blockIdx.y * 256 + wave * 32;

    const int scol  = tid >> 1;
    const int skoff = (tid & 1) * 32;
    const int8_t* bsrc = wq + (size_t)(n0 + scol) * Kdim + skoff;
    const unsigned bdst0 = lds_addr_of(&bsm[0][scol * LDS_STRIDE + skoff]);
    const unsigned bdst1 = lds_addr_of(&bsm[1][scol * LDS_STRIDE + skoff]);

    const int nsteps = Kdim >> 6;

    async_ld_b128(bdst0,      bsrc);
    async_ld_b128(bdst0 + 16, bsrc + 16);
    if (nsteps > 1) {
        async_ld_b128(bdst1,      bsrc + 64);
        async_ld_b128(bdst1 + 16, bsrc + 80);
    }

    const int a_koff = (lane >> 4) * 8;
    const int8_t* arow0 = xq + (size_t)(m0 + (lane & 15)) * Kdim + a_koff;
    const int8_t* arow1 = arow0 + (size_t)16 * Kdim;

    U8 a_cur[2], a_next[2];
    {
        const v2i* p0 = (const v2i*)arow0;
        a_cur[0].d[0] = p0[0]; a_cur[0].d[1] = p0[2];
        a_cur[0].d[2] = p0[4]; a_cur[0].d[3] = p0[6];
        const v2i* p1 = (const v2i*)arow1;
        a_cur[1].d[0] = p1[0]; a_cur[1].d[1] = p1[2];
        a_cur[1].d[2] = p1[4]; a_cur[1].d[3] = p1[6];
    }

    v8i acc[2][8];
#pragma unroll
    for (int u = 0; u < 2; ++u)
#pragma unroll
        for (int t = 0; t < 8; ++t) acc[u][t] = v8i{0, 0, 0, 0, 0, 0, 0, 0};

    const int frag_off = (lane & 15) * LDS_STRIDE + (lane >> 4) * 8;

    for (int s = 0; s < nsteps; ++s) {
        if (s + 1 < nsteps) {
            const size_t ko = (size_t)(s + 1) * 64;
            const v2i* p0 = (const v2i*)(arow0 + ko);
            a_next[0].d[0] = p0[0]; a_next[0].d[1] = p0[2];
            a_next[0].d[2] = p0[4]; a_next[0].d[3] = p0[6];
            const v2i* p1 = (const v2i*)(arow1 + ko);
            a_next[1].d[0] = p1[0]; a_next[1].d[1] = p1[2];
            a_next[1].d[2] = p1[4]; a_next[1].d[3] = p1[6];
        }

        if (s + 1 < nsteps) wait_async_le2();
        else                wait_async_le0();
        __syncthreads();

        const int8_t* bb = &bsm[s & 1][frag_off];
#pragma unroll
        for (int t = 0; t < 8; ++t) {
            const int8_t* p = bb + t * 16 * LDS_STRIDE;
            U8 bf;
            bf.d[0] = *(const v2i*)(p);
            bf.d[1] = *(const v2i*)(p + 16);
            bf.d[2] = *(const v2i*)(p + 32);
            bf.d[3] = *(const v2i*)(p + 48);
            acc[0][t] = __builtin_amdgcn_wmma_i32_16x16x64_iu8(
                true, a_cur[0].v, true, bf.v, acc[0][t], false, false);
            acc[1][t] = __builtin_amdgcn_wmma_i32_16x16x64_iu8(
                true, a_cur[1].v, true, bf.v, acc[1][t], false, false);
        }

        __syncthreads();

        if (s + 2 < nsteps) {
            const int8_t* g = bsrc + (size_t)(s + 2) * 64;
            const unsigned d = (s & 1) ? bdst1 : bdst0;
            async_ld_b128(d,      g);
            async_ld_b128(d + 16, g + 16);
        }
        if (s + 1 < nsteps) {
            a_cur[0] = a_next[0];
            a_cur[1] = a_next[1];
        }
    }

    const float s_x = __uint_as_float(*absmax_bits) * (1.0f / 127.0f);
    const int orow = (lane >> 4) * 8;
    const int ocol = lane & 15;
#pragma unroll
    for (int t = 0; t < 8; ++t) {
        const int n      = n0 + t * 16 + ocol;
        const float scal = s_x * sw[n];
        const float bi   = bias[n];
#pragma unroll
        for (int u = 0; u < 2; ++u) {
#pragma unroll
            for (int r = 0; r < 8; ++r) {
                const int m = m0 + u * 16 + orow + r;
                out[(size_t)m * Ndim + n] = (float)acc[u][t][r] * scal + bi;
            }
        }
    }
}

extern "C" void kernel_launch(void* const* d_in, const int* in_sizes, int n_in,
                              void* d_out, int out_size, void* d_ws, size_t ws_size,
                              hipStream_t stream) {
    const float*  x    = (const float*)d_in[0];
    const int8_t* wq   = (const int8_t*)d_in[1];
    const float*  sw   = (const float*)d_in[2];
    const float*  bias = (const float*)d_in[3];
    float*        out  = (float*)d_out;

    const long long Ndim = in_sizes[2];
    const long long Kdim = (long long)in_sizes[1] / Ndim;
    const long long Mdim = (long long)in_sizes[0] / Kdim;

    unsigned* amax = (unsigned*)d_ws;
    int8_t*   xq   = (int8_t*)d_ws + 256;

    const long long n4 = (Mdim * Kdim) / 4;

    ws_init_kernel<<<1, 1, 0, stream>>>(amax);
    absmax_kernel<<<2048, 256, 0, stream>>>((const float4*)x, amax, n4);
    quant_kernel<<<4096, 256, 0, stream>>>((const float4*)x, (int*)xq, amax, n4);

    dim3 grid((unsigned)(Ndim / 128), (unsigned)(Mdim / 256));
    gemm_w8a8_kernel<<<grid, 256, 0, stream>>>(xq, wq, sw, bias, amax, out,
                                               (int)Mdim, (int)Ndim, (int)Kdim);
}
